// BalSupMoCoNet_47708496724526
// MI455X (gfx1250) — matrix operand order repaired and checked by reading, not compiled, not run on hardware
//
#include <hip/hip_runtime.h>

// ---------------- problem constants ----------------
#define TEMP_INV 14.285714285714285714f     // 1/0.07 ; exact softmax stabilizer (shift-invariant)
#define CE2     20.609929155556625f         // log2(e)/T : exp((c-1)/T) == exp2(c*CE2 - CE2)
constexpr int B_ = 256;
constexpr int K_ = 8192;
constexpr int D_ = 128;
constexpr int N_ = B_ + 2 * K_;             // 16640
constexpr int ROWS_PER_BLOCK = 128;         // 8 waves * 16 rows
constexpr int JSPLIT = 8;                   // j-chunks, combined deterministically
constexpr int JTILES_PER = (N_ / 16) / JSPLIT;  // 130 tiles of 16 columns (even)

// ---------------- WMMA types ----------------
typedef __attribute__((ext_vector_type(16))) __bf16 v16bf;
typedef __attribute__((ext_vector_type(8)))  float  v8f;

union Frag { v16bf v; uint4 q[2]; };  // 32 bytes = 16 bf16 halves

#if __has_builtin(__builtin_amdgcn_exp2f)
#define EXP2F(x) __builtin_amdgcn_exp2f(x)
#else
#define EXP2F(x) __expf((x) * 0.6931471805599453f)
#endif

__device__ __forceinline__ unsigned short f2bf(float f) {
  unsigned int u = __float_as_uint(f);
  u = (u + 0x7FFFu + ((u >> 16) & 1u)) >> 16;   // round-to-nearest-even
  return (unsigned short)u;
}
__device__ __forceinline__ float bf2f(unsigned short h) {
  return __uint_as_float(((unsigned int)h) << 16);
}

// load one 16x16x32 B fragment: column = jrow, K-half by lane half
__device__ __forceinline__ void loadB(Frag& b, const unsigned short* pb, int kc) {
  b.q[0] = *(const uint4*)(pb + kc * 32);
  b.q[1] = *(const uint4*)(pb + kc * 32 + 8);
}

// ---------------- kernel 1: normalize q -> bf16 features, labels[0:256] ----------------
__global__ void normq_kernel(const float* __restrict__ q,
                             const int* __restrict__ targets,
                             unsigned short* __restrict__ F,
                             int* __restrict__ labels) {
  __shared__ float sm[D_];
  const int row = blockIdx.x, t = threadIdx.x;
  const float x = q[row * D_ + t];
  sm[t] = x * x;
  __syncthreads();
  for (int s = D_ / 2; s > 0; s >>= 1) {
    if (t < s) sm[t] += sm[t + s];
    __syncthreads();
  }
  const float nrm = fmaxf(sqrtf(sm[0]), 1e-12f);
  F[row * D_ + t] = f2bf(x / nrm);
  if (t == 0) labels[row] = targets[row];
}

// ---------------- kernel 2: queues -> bf16, labels ----------------
__global__ void prep_kernel(const float* __restrict__ ba,
                            const float* __restrict__ nba,
                            unsigned short* __restrict__ F,
                            int* __restrict__ labels) {
  const long QN = (long)K_ * D_;
  const long stride = (long)gridDim.x * blockDim.x;
  for (long idx = (long)blockIdx.x * blockDim.x + threadIdx.x; idx < 2 * QN; idx += stride) {
    float v; int row, col;
    if (idx < QN) {
      v = ba[idx];  row = B_ + (int)(idx / D_);        col = (int)(idx % D_);
    } else {
      long j = idx - QN;
      v = nba[j];   row = B_ + K_ + (int)(j / D_);     col = (int)(j % D_);
    }
    F[row * D_ + col] = f2bf(v);
    if (col == 0) labels[row] = (idx < QN) ? 1 : 0;
  }
}

// ---------------- kernel 3: class-sum vectors G[2][128] ----------------
__global__ void gsum_kernel(const unsigned short* __restrict__ F,
                            const int* __restrict__ labels,
                            float* __restrict__ G) {
  const int k = blockIdx.x;        // column 0..127
  const int t = threadIdx.x;       // 256 threads
  float s0 = 0.0f, s1 = 0.0f;
  for (int i = t; i < N_; i += 256) {
    const float x = bf2f(F[i * D_ + k]);
    if (labels[i]) s1 += x; else s0 += x;
  }
  __shared__ float a0[256], a1[256];
  a0[t] = s0; a1[t] = s1;
  __syncthreads();
  for (int s = 128; s > 0; s >>= 1) {
    if (t < s) { a0[t] += a0[t + s]; a1[t] += a1[t + s]; }
    __syncthreads();
  }
  if (t == 0) { G[k] = a0[0]; G[D_ + k] = a1[0]; }
}

// ---------------- kernel 4: Gram tiles + streaming exp-sum (branch-free hot loop) --------
__global__ void __launch_bounds__(256)
sim_kernel(const unsigned short* __restrict__ F,
           float* __restrict__ Spart /* [JSPLIT][N] */) {
  const int lane = threadIdx.x & 31;
  const int wave = threadIdx.x >> 5;
  const int hi   = lane >> 4;         // lane half (0/1)
  const int ln   = lane & 15;
  const int ibase = blockIdx.x * ROWS_PER_BLOCK + wave * 16;

  // --- A fragments: rows [ibase, ibase+16), K=128 in 4 chunks of 32 (contiguous k = 16*half + s, same as B) ---
  const unsigned short* pa = F + (ibase + ln) * D_;
  Frag a[4];
#pragma unroll
  for (int kc = 0; kc < 4; ++kc) {
    const unsigned short* p = pa + kc * 32 + hi * 16;  // lanes 0-15: K 0..15 ; 16-31: K 16..31
    a[kc].q[0] = *(const uint4*)(p);                   // halves 0..7  (K 0..7 / 16..23)
    a[kc].q[1] = *(const uint4*)(p + 8);               // halves 8..15 (K 8..15 / 24..31)
  }

  float esum[8];
#pragma unroll
  for (int r = 0; r < 8; ++r) esum[r] = 0.0f;

  const int jstart = blockIdx.y * JTILES_PER * 16;
  const unsigned short* pb0 = F + (jstart + ln) * D_ + hi * 16;

  // two j-tiles per iteration: 16 independent b128 loads + 8 WMMAs of ILP
  for (int jt = 0; jt < JTILES_PER; jt += 2) {
    const unsigned short* pb1 = pb0 + 16 * D_;
    Frag b00, b01, b02, b03, b10, b11, b12, b13;
    loadB(b00, pb0, 0); loadB(b01, pb0, 1); loadB(b02, pb0, 2); loadB(b03, pb0, 3);
    loadB(b10, pb1, 0); loadB(b11, pb1, 1); loadB(b12, pb1, 2); loadB(b13, pb1, 3);
    __builtin_prefetch(pb0 + 32 * D_, 0, 3);           // next pair (stays inside ws)

    v8f c0 = {0.f, 0.f, 0.f, 0.f, 0.f, 0.f, 0.f, 0.f};
    v8f c1 = {0.f, 0.f, 0.f, 0.f, 0.f, 0.f, 0.f, 0.f};
    c0 = __builtin_amdgcn_wmma_f32_16x16x32_bf16(false, a[0].v, false, b00.v, (short)0, c0, false, false);
    c0 = __builtin_amdgcn_wmma_f32_16x16x32_bf16(false, a[1].v, false, b01.v, (short)0, c0, false, false);
    c0 = __builtin_amdgcn_wmma_f32_16x16x32_bf16(false, a[2].v, false, b02.v, (short)0, c0, false, false);
    c0 = __builtin_amdgcn_wmma_f32_16x16x32_bf16(false, a[3].v, false, b03.v, (short)0, c0, false, false);
    c1 = __builtin_amdgcn_wmma_f32_16x16x32_bf16(false, a[0].v, false, b10.v, (short)0, c1, false, false);
    c1 = __builtin_amdgcn_wmma_f32_16x16x32_bf16(false, a[1].v, false, b11.v, (short)0, c1, false, false);
    c1 = __builtin_amdgcn_wmma_f32_16x16x32_bf16(false, a[2].v, false, b12.v, (short)0, c1, false, false);
    c1 = __builtin_amdgcn_wmma_f32_16x16x32_bf16(false, a[3].v, false, b13.v, (short)0, c1, false, false);

    // exp((sim-1)/T) = exp2(c*CE2 - CE2): one FMA + one v_exp per element, no masks
#pragma unroll
    for (int r = 0; r < 8; ++r) {
      esum[r] += EXP2F(__builtin_fmaf(c0[r], CE2, -CE2));
      esum[r] += EXP2F(__builtin_fmaf(c1[r], CE2, -CE2));
    }
    pb0 += 32 * D_;
  }

  // --- diagonal correction: only the chunk containing columns [ibase, ibase+16) ---
  if (ibase >= jstart && ibase < jstart + JTILES_PER * 16) {
    const unsigned short* pbd = F + (ibase + ln) * D_ + hi * 16;
    Frag d0, d1, d2, d3;
    loadB(d0, pbd, 0); loadB(d1, pbd, 1); loadB(d2, pbd, 2); loadB(d3, pbd, 3);
    v8f cd = {0.f, 0.f, 0.f, 0.f, 0.f, 0.f, 0.f, 0.f};
    cd = __builtin_amdgcn_wmma_f32_16x16x32_bf16(false, a[0].v, false, d0.v, (short)0, cd, false, false);
    cd = __builtin_amdgcn_wmma_f32_16x16x32_bf16(false, a[1].v, false, d1.v, (short)0, cd, false, false);
    cd = __builtin_amdgcn_wmma_f32_16x16x32_bf16(false, a[2].v, false, d2.v, (short)0, cd, false, false);
    cd = __builtin_amdgcn_wmma_f32_16x16x32_bf16(false, a[3].v, false, d3.v, (short)0, cd, false, false);
#pragma unroll
    for (int r = 0; r < 8; ++r) {
      if (r + 8 * hi == ln)                                  // this lane holds row==col element
        esum[r] -= EXP2F(__builtin_fmaf(cd[r], CE2, -CE2));  // remove the self term
    }
  }

  // --- fold 16 lanes (one tile-row each); deterministic per-(chunk,row) slot, no atomics ---
#pragma unroll
  for (int r = 0; r < 8; ++r) {
    float e = esum[r];
#pragma unroll
    for (int m = 8; m >= 1; m >>= 1) e += __shfl_xor(e, m, 16);
    if (ln == 0) {
      const int ri = ibase + r + 8 * hi;
      Spart[blockIdx.y * N_ + ri] = e;
    }
  }
}

// ---------------- kernel 5: per-row log-prob ----------------
__global__ void rowlp_kernel(const unsigned short* __restrict__ F,
                             const int* __restrict__ labels,
                             const float* __restrict__ Spart,
                             const float* __restrict__ G,
                             float* __restrict__ rowLP) {
  __shared__ float gs[2 * D_];
  __shared__ int   si[D_];
  const int t = threadIdx.x;            // 128 threads
  gs[t] = G[t]; gs[D_ + t] = G[D_ + t];
  si[t] = labels[t] + labels[t + D_];   // targets occupy labels[0:256]
  __syncthreads();
  for (int s = 64; s > 0; s >>= 1) {
    if (t < s) si[t] += si[t + s];
    __syncthreads();
  }
  const int t1 = si[0];
  const float cnt1 = (float)(K_ + t1) - 1.0f;
  const float cnt0 = (float)(K_ + B_ - t1) - 1.0f;

  const int row = blockIdx.x * D_ + t;
  const int lbl = labels[row];
  const unsigned short* f = F + row * D_;
  const float* g = gs + lbl * D_;
  float dg = 0.0f, n2 = 0.0f;
#pragma unroll 8
  for (int k = 0; k < D_; ++k) {
    const float x = bf2f(f[k]);
    dg += x * g[k];
    n2 += x * x;
  }
  float S = 0.0f;
#pragma unroll
  for (int cch = 0; cch < JSPLIT; ++cch) S += Spart[cch * N_ + row];

  const float P   = (dg - n2) * TEMP_INV;          // sum of positive sims, self excluded
  const float cnt = lbl ? cnt1 : cnt0;
  rowLP[row] = P / cnt - TEMP_INV - logf(S);
}

// ---------------- kernel 6: mean ----------------
__global__ void reduce_kernel(const float* __restrict__ rowLP, float* __restrict__ out) {
  __shared__ float sm[256];
  const int t = threadIdx.x;
  float acc = 0.0f;
  for (int i = t; i < N_; i += 256) acc += rowLP[i];
  sm[t] = acc;
  __syncthreads();
  for (int s = 128; s > 0; s >>= 1) {
    if (t < s) sm[t] += sm[t + s];
    __syncthreads();
  }
  if (t == 0) out[0] = -sm[0] / (float)N_;
}

// ---------------- launcher ----------------
extern "C" void kernel_launch(void* const* d_in, const int* in_sizes, int n_in,
                              void* d_out, int out_size, void* d_ws, size_t ws_size,
                              hipStream_t stream) {
  const float* q       = (const float*)d_in[0];
  const float* ba      = (const float*)d_in[1];
  const float* nba     = (const float*)d_in[2];
  const int*   targets = (const int*)d_in[3];

  char* ws = (char*)d_ws;
  unsigned short* F = (unsigned short*)ws;                       // N*128 bf16 = 4,259,840 B
  size_t off = (size_t)N_ * D_ * sizeof(unsigned short);
  int*   labels = (int*)(ws + off);      off += (size_t)N_ * 4;  // N int32
  float* Spart  = (float*)(ws + off);    off += (size_t)JSPLIT * N_ * 4;  // 8*N f32
  float* G      = (float*)(ws + off);    off += 2 * D_ * 4;      // 2*128 f32
  float* rowLP  = (float*)(ws + off);                            // N f32

  normq_kernel<<<B_, D_, 0, stream>>>(q, targets, F, labels);
  prep_kernel<<<2048, 256, 0, stream>>>(ba, nba, F, labels);
  gsum_kernel<<<D_, 256, 0, stream>>>(F, labels, G);

  dim3 grid(N_ / ROWS_PER_BLOCK, JSPLIT);   // 130 x 8 = 1040 workgroups
  sim_kernel<<<grid, 256, 0, stream>>>(F, Spart);

  rowlp_kernel<<<N_ / D_, D_, 0, stream>>>(F, labels, Spart, G, rowLP);
  reduce_kernel<<<1, 256, 0, stream>>>(rowLP, (float*)d_out);
}
